// HSauteUnit_57028575756931
// MI455X (gfx1250) — compile-verified, matrix operand order checked by reading, not run on hardware
//
#include <hip/hip_runtime.h>
#include <hip/hip_bf16.h>

// ---------------------------------------------------------------------------
// Model dims (fixed by the reference)
// ---------------------------------------------------------------------------
#define MB   16     // batch (scan length)
#define MT   512    // tokens
#define MD   768    // model dim
#define MH   12     // heads
#define MHD  64     // head dim
#define MF   3072   // ffn dim
#define ML   2      // layers
#define MS   8      // speakers
#define MU   2048   // gate hidden
#define MD3  2304   // 3*D

// ---------------------------------------------------------------------------
// Types for WMMA
// ---------------------------------------------------------------------------
typedef __bf16  bf16x16 __attribute__((ext_vector_type(16)));
typedef float   f32x8   __attribute__((ext_vector_type(8)));

union Frag {
    uint4   q[2];
    bf16x16 v;
};

__device__ __forceinline__ unsigned short f2bf(float f) {
    unsigned u = __float_as_uint(f);
    u += 0x7FFFu + ((u >> 16) & 1u);        // round-to-nearest-even
    return (unsigned short)(u >> 16);
}

__device__ __forceinline__ unsigned pack2bf(float lo, float hi) {
    return (unsigned)f2bf(lo) | ((unsigned)f2bf(hi) << 16);
}

__device__ __forceinline__ float gelu_f(float x) {
    return 0.5f * x * (1.0f + erff(x * 0.70710678118654752f));
}

// ---------------------------------------------------------------------------
// Tile staging helpers.  Tiles: A 128x32 (row-major m,k), B 64x32 stored
// transposed in LDS as (n,k).  256 threads.
//   A: fp32, K-contiguous (saK==1 for every GEMM in this model).
//   B modes: 2 = bf16 pre-transposed [n][k] (weights)  -> pure b128 copy
//            1 = fp32, K-contiguous columns (scores' K^T view)
//            0 = fp32 row-major [k][n], N-contiguous (A·V's V view)
// ---------------------------------------------------------------------------
__device__ __forceinline__ void stage_tileA(unsigned short* __restrict__ dst,
                                            const float* __restrict__ A,
                                            long saM, long bM, int kc, int tid)
{
    #pragma unroll
    for (int s = tid; s < (128 * 32) / 8; s += 256) {
        const int row = s >> 2, seg = s & 3;
        const float4* gp = (const float4*)(A + (bM + row) * saM + kc + seg * 8);
        const float4 p0 = gp[0], p1 = gp[1];
        uint4 w;
        w.x = pack2bf(p0.x, p0.y); w.y = pack2bf(p0.z, p0.w);
        w.z = pack2bf(p1.x, p1.y); w.w = pack2bf(p1.z, p1.w);
        *(uint4*)(dst + row * 32 + seg * 8) = w;
    }
}

template<int BMODE>
__device__ __forceinline__ void stage_tileB(unsigned short* __restrict__ dst,
                                            const void* __restrict__ Bv,
                                            long sbStride, long bN, int kc, int tid)
{
    if constexpr (BMODE == 2) {
        const int n = tid >> 2, seg = tid & 3;
        const unsigned short* Bt = (const unsigned short*)Bv;
        const uint4* gp = (const uint4*)(Bt + (bN + n) * sbStride + kc + seg * 8);
        *(uint4*)(dst + n * 32 + seg * 8) = gp[0];
    } else if constexpr (BMODE == 1) {
        const int n = tid >> 2, seg = tid & 3;
        const float* B = (const float*)Bv;
        const float4* gp = (const float4*)(B + (bN + n) * sbStride + kc + seg * 8);
        const float4 p0 = gp[0], p1 = gp[1];
        uint4 w;
        w.x = pack2bf(p0.x, p0.y); w.y = pack2bf(p0.z, p0.w);
        w.z = pack2bf(p1.x, p1.y); w.w = pack2bf(p1.z, p1.w);
        *(uint4*)(dst + n * 32 + seg * 8) = w;
    } else {
        const int k = tid >> 3, n0 = (tid & 7) * 8;
        const float* B = (const float*)Bv;
        const float4* gp = (const float4*)(B + (long)(kc + k) * sbStride + bN + n0);
        const float4 p0 = gp[0], p1 = gp[1];
        const float v[8] = {p0.x, p0.y, p0.z, p0.w, p1.x, p1.y, p1.z, p1.w};
        #pragma unroll
        for (int j = 0; j < 8; ++j) dst[(n0 + j) * 32 + k] = f2bf(v[j]);
    }
}

// ---------------------------------------------------------------------------
// WMMA GEMM: C[z] = epi( scale * A[z](MxK) @ B[z](KxN) )
// Block 128x64, BK=32, double-buffered LDS, 8 waves each computing 32x32
// (4 x v_wmma_f32_16x16x32_bf16 per K-chunk).
// EPI: 0 = scale only, 1 = +bias, 2 = +bias then GELU, 3 = +bias +residual
// Requires M%128==0, N%64==0, K%32==0 (true for every call in this model).
// ---------------------------------------------------------------------------
template<int EPI, int BMODE>
__global__ __launch_bounds__(256)
void wmma_gemm_kernel(const float* __restrict__ A, long saM, long saZ,
                      const void* __restrict__ Bv, long sbStride, long sbZ,
                      float* __restrict__ C, long scM, long scZ,
                      const float* __restrict__ bias,
                      const float* __restrict__ resid,
                      int K, float scale)
{
    __shared__ __align__(16) unsigned short As[2][128 * 32];
    __shared__ __align__(16) unsigned short Bs[2][64 * 32];

    const int  tid = threadIdx.x;
    const int  z   = blockIdx.z;
    const long bM  = (long)blockIdx.y * 128;
    const long bN  = (long)blockIdx.x * 64;

    A += (long)z * saZ;
    C += (long)z * scZ;
    const void* Bz = (BMODE == 2)
        ? (const void*)((const unsigned short*)Bv + (long)z * sbZ)
        : (const void*)((const float*)Bv + (long)z * sbZ);

    const int wid   = tid >> 5;
    const int lane  = tid & 31;
    const int half  = lane >> 4;   // 0: lanes 0-15, 1: lanes 16-31
    const int r     = lane & 15;
    const int waveM = wid >> 1;    // 0..3 -> 32-row slices
    const int waveN = wid & 1;     // 0..1 -> 32-col slices

    f32x8 acc00 = {}, acc01 = {}, acc10 = {}, acc11 = {};

    stage_tileA(As[0], A, saM, bM, 0, tid);
    stage_tileB<BMODE>(Bs[0], Bz, sbStride, bN, 0, tid);
    __syncthreads();

    int buf = 0;
    for (int kc = 0; kc < K; kc += 32) {
        if (kc + 32 < K) {
            stage_tileA(As[buf ^ 1], A, saM, bM, kc + 32, tid);
            stage_tileB<BMODE>(Bs[buf ^ 1], Bz, sbStride, bN, kc + 32, tid);
            if (kc + 64 < K)   // one chunk further ahead (global_prefetch_b8)
                __builtin_prefetch(A + (bM + (tid & 127)) * saM + kc + 64, 0, 1);
        }

        // fragments: lane half 0 -> K {0..7}+{16..23}; half 1 -> {8..15}+{24..31}
        Frag a0, a1, b0, b1;
        const unsigned short* as = As[buf];
        const unsigned short* bs = Bs[buf];
        { const uint4* p = (const uint4*)(as + (waveM * 32 +      r) * 32 + half * 8); a0.q[0] = p[0]; a0.q[1] = p[2]; }
        { const uint4* p = (const uint4*)(as + (waveM * 32 + 16 + r) * 32 + half * 8); a1.q[0] = p[0]; a1.q[1] = p[2]; }
        { const uint4* p = (const uint4*)(bs + (waveN * 32 +      r) * 32 + half * 8); b0.q[0] = p[0]; b0.q[1] = p[2]; }
        { const uint4* p = (const uint4*)(bs + (waveN * 32 + 16 + r) * 32 + half * 8); b1.q[0] = p[0]; b1.q[1] = p[2]; }

        acc00 = __builtin_amdgcn_wmma_f32_16x16x32_bf16(false, a0.v, false, b0.v, (short)0, acc00, false, false);
        acc01 = __builtin_amdgcn_wmma_f32_16x16x32_bf16(false, a0.v, false, b1.v, (short)0, acc01, false, false);
        acc10 = __builtin_amdgcn_wmma_f32_16x16x32_bf16(false, a1.v, false, b0.v, (short)0, acc10, false, false);
        acc11 = __builtin_amdgcn_wmma_f32_16x16x32_bf16(false, a1.v, false, b1.v, (short)0, acc11, false, false);
        __syncthreads();
        buf ^= 1;
    }

    // epilogue: C/D layout -> lane (half*8 + i) is M-sub, r is N-sub
    #pragma unroll
    for (int ai = 0; ai < 2; ++ai) {
        const f32x8 aa0 = ai ? acc10 : acc00;
        const f32x8 aa1 = ai ? acc11 : acc01;
        #pragma unroll
        for (int i = 0; i < 8; ++i) {
            const long m  = bM + waveM * 32 + ai * 16 + half * 8 + i;
            const long n0 = bN + waveN * 32 + r;
            const long n1 = n0 + 16;
            float v0 = aa0[i] * scale;
            float v1 = aa1[i] * scale;
            if constexpr (EPI >= 1) { v0 += bias[n0]; v1 += bias[n1]; }
            if constexpr (EPI == 2) { v0 = gelu_f(v0); v1 = gelu_f(v1); }
            if constexpr (EPI == 3) { v0 += resid[m * scM + n0]; v1 += resid[m * scM + n1]; }
            C[m * scM + n0] = v0;
            C[m * scM + n1] = v1;
        }
    }
}

// ---------------------------------------------------------------------------
// One-time weight convert+transpose: Wt[n*K + k] = bf16(W[k*N + n])
// ---------------------------------------------------------------------------
__global__ void transpose_bf16_kernel(const float* __restrict__ W,
                                      unsigned short* __restrict__ Wt,
                                      int K, int N)
{
    const long i = (long)blockIdx.x * 256 + threadIdx.x;   // over N*K, (n,k)
    if (i >= (long)K * N) return;
    const int n = (int)(i / K), k = (int)(i % K);
    Wt[i] = f2bf(W[(long)k * N + n]);
}

// ---------------------------------------------------------------------------
// Small helper kernels (speaker path / normalization / embedding)
// ---------------------------------------------------------------------------
__global__ void embed_kernel(const int* __restrict__ ids,
                             const float* __restrict__ tok,
                             const float* __restrict__ pos,
                             float* __restrict__ out)
{
    const long bt = blockIdx.x;                  // b*T + t
    const int  id = ids[bt];
    const int  t  = (int)(bt % MT);
    const int  p  = (id != 0) ? t : 0;
    for (int d = threadIdx.x; d < MD; d += 256)
        out[bt * MD + d] = tok[(long)id * MD + d] + pos[(long)p * MD + d];
}

__global__ void zero_kernel(float* p, int n)
{
    const int i = blockIdx.x * 256 + threadIdx.x;
    if (i < n) p[i] = 0.0f;
}

__global__ void add_rowvec_kernel(const float* __restrict__ e,
                                  const float* __restrict__ v,
                                  float* __restrict__ out)
{
    const long row = blockIdx.x;
    for (int d = threadIdx.x; d < MD; d += 256)
        out[row * MD + d] = e[row * MD + d] + v[d];
}

// out[n] = act( a . W[:,n] + bias[n] ), W row-major KxN.  sel: optional row select.
template<int ACT>
__global__ void vecmat_kernel(const float* __restrict__ a,
                              const float* __restrict__ W,
                              const float* __restrict__ bias,
                              float* __restrict__ out,
                              int K, int N,
                              const int* __restrict__ sel, int selIdx)
{
    const int n = blockIdx.x * 256 + threadIdx.x;
    if (n >= N) return;
    if (sel) a += (long)sel[selIdx] * K;
    float s = bias ? bias[n] : 0.0f;
    for (int k = 0; k < K; ++k) s += a[k] * W[(long)k * N + n];
    if constexpr (ACT == 1) s = gelu_f(s);
    out[n] = s;
}

__global__ void softmax_rows_kernel(float* __restrict__ x, int N)
{
    __shared__ float red[256];
    float* xr = x + (long)blockIdx.x * N;
    const int tid = threadIdx.x;

    float mx = -1e30f;
    for (int i = tid; i < N; i += 256) mx = fmaxf(mx, xr[i]);
    red[tid] = mx; __syncthreads();
    for (int o = 128; o > 0; o >>= 1) {
        if (tid < o) red[tid] = fmaxf(red[tid], red[tid + o]);
        __syncthreads();
    }
    mx = red[0]; __syncthreads();

    float s = 0.0f;
    for (int i = tid; i < N; i += 256) { float e = __expf(xr[i] - mx); xr[i] = e; s += e; }
    red[tid] = s; __syncthreads();
    for (int o = 128; o > 0; o >>= 1) {
        if (tid < o) red[tid] += red[tid + o];
        __syncthreads();
    }
    const float inv = 1.0f / red[0];
    for (int i = tid; i < N; i += 256) xr[i] *= inv;
}

__global__ void layernorm_kernel(const float* __restrict__ x,
                                 const float* __restrict__ g,
                                 const float* __restrict__ bta,
                                 float* __restrict__ out)
{
    __shared__ float red[256];
    const float* xr = x + (long)blockIdx.x * MD;
    const int tid = threadIdx.x;

    float s = 0.0f;
    for (int d = tid; d < MD; d += 256) s += xr[d];
    red[tid] = s; __syncthreads();
    for (int o = 128; o > 0; o >>= 1) { if (tid < o) red[tid] += red[tid + o]; __syncthreads(); }
    const float mean = red[0] / MD; __syncthreads();

    float v = 0.0f;
    for (int d = tid; d < MD; d += 256) { float t = xr[d] - mean; v += t * t; }
    red[tid] = v; __syncthreads();
    for (int o = 128; o > 0; o >>= 1) { if (tid < o) red[tid] += red[tid + o]; __syncthreads(); }
    const float rstd = rsqrtf(red[0] / MD + 1e-5f);

    for (int d = tid; d < MD; d += 256)
        out[(long)blockIdx.x * MD + d] = (xr[d] - mean) * rstd * g[d] + bta[d];
}

__global__ void masked_mean_kernel(const float* __restrict__ x,
                                   const int* __restrict__ ids,
                                   float* __restrict__ u)
{
    const int d = blockIdx.x * 256 + threadIdx.x;
    if (d >= MD) return;
    float s = 0.0f, c = 0.0f;
    for (int t = 0; t < MT; ++t) {
        const float m = (ids[t] != 0) ? 1.0f : 0.0f;
        s += x[(long)t * MD + d] * m;
        c += m;
    }
    u[d] = s / fmaxf(c, 1e-6f);
}

// logits = h_spk @ uW ; w = softmax(logits) ; ctx = w @ h_spk ;
// mat = concat(u, h_spk[sid], ctx)
__global__ void spk_attn_kernel(const float* __restrict__ hspk,
                                const float* __restrict__ uW,
                                const float* __restrict__ u,
                                const int* __restrict__ sids, int b,
                                float* __restrict__ mat)
{
    __shared__ float logits[MS];
    __shared__ float wsm[MS];
    const int tid = threadIdx.x, wid = tid >> 5, lane = tid & 31;

    if (wid < MS) {
        float s = 0.0f;
        for (int k = lane; k < MD; k += 32) s += hspk[(long)wid * MD + k] * uW[k];
        for (int o = 16; o > 0; o >>= 1) s += __shfl_down(s, o);
        if (lane == 0) logits[wid] = s;
    }
    __syncthreads();
    if (tid == 0) {
        float mx = -1e30f;
        for (int s = 0; s < MS; ++s) mx = fmaxf(mx, logits[s]);
        float tot = 0.0f;
        for (int s = 0; s < MS; ++s) { float e = __expf(logits[s] - mx); wsm[s] = e; tot += e; }
        for (int s = 0; s < MS; ++s) wsm[s] /= tot;
    }
    __syncthreads();
    const int sid = sids[b];
    for (int d = tid; d < MD; d += 256) {
        float c = 0.0f;
        for (int s = 0; s < MS; ++s) c += wsm[s] * hspk[(long)s * MD + d];
        mat[d]          = u[d];
        mat[MD + d]     = hspk[(long)sid * MD + d];
        mat[2 * MD + d] = c;
    }
}

// h_spk[sid] += LayerNorm(nh) * g + b
__global__ void ln_add_hspk_kernel(const float* __restrict__ nh,
                                   const float* __restrict__ g,
                                   const float* __restrict__ bta,
                                   float* __restrict__ hspk,
                                   const int* __restrict__ sids, int b)
{
    __shared__ float red[256];
    const int tid = threadIdx.x;

    float s = 0.0f;
    for (int d = tid; d < MD; d += 256) s += nh[d];
    red[tid] = s; __syncthreads();
    for (int o = 128; o > 0; o >>= 1) { if (tid < o) red[tid] += red[tid + o]; __syncthreads(); }
    const float mean = red[0] / MD; __syncthreads();

    float v = 0.0f;
    for (int d = tid; d < MD; d += 256) { float t = nh[d] - mean; v += t * t; }
    red[tid] = v; __syncthreads();
    for (int o = 128; o > 0; o >>= 1) { if (tid < o) red[tid] += red[tid + o]; __syncthreads(); }
    const float rstd = rsqrtf(red[0] / MD + 1e-5f);

    const int sid = sids[b];
    for (int d = tid; d < MD; d += 256)
        hspk[(long)sid * MD + d] += (nh[d] - mean) * rstd * g[d] + bta[d];
}

// ---------------------------------------------------------------------------
// Host launcher
// ---------------------------------------------------------------------------
extern "C" void kernel_launch(void* const* d_in, const int* in_sizes, int n_in,
                              void* d_out, int out_size, void* d_ws, size_t ws_size,
                              hipStream_t stream)
{
    (void)in_sizes; (void)n_in; (void)out_size; (void)ws_size;

    const int*   input_ids   = (const int*)  d_in[0];
    const int*   speaker_ids = (const int*)  d_in[1];
    const float* tok_emb     = (const float*)d_in[2];
    const float* pos_emb     = (const float*)d_in[3];

    float* out = (float*)d_out;      // (B, T, D) — the scan-carried emb
    float* ws  = (float*)d_ws;

    // fp32 workspace carve
    size_t off = 0;
    float* x0      = ws + off; off += (size_t)MT * MD;   // residual input
    float* x1      = ws + off; off += (size_t)MT * MD;   // post-LN1
    float* qkv     = ws + off; off += (size_t)MT * MD3;
    float* scores  = ws + off; off += (size_t)MH * MT * MT;
    float* attnout = ws + off; off += (size_t)MT * MD;
    float* hbuf    = ws + off; off += (size_t)MT * MF;
    float* tmp     = ws + off; off += (size_t)MT * MD;
    float* h_spk   = ws + off; off += (size_t)MS * MD;
    float* proj    = ws + off; off += MD;
    float* u       = ws + off; off += MD;
    float* uW      = ws + off; off += MD;
    float* mat     = ws + off; off += MD3;
    float* gbuf    = ws + off; off += MU;
    float* nh      = ws + off; off += MD;

    // bf16 pre-transposed weight region (16B aligned)
    unsigned long long pa = (unsigned long long)(ws + off);
    pa = (pa + 15ull) & ~15ull;
    unsigned short* bfp   = (unsigned short*)pa;
    unsigned short* wqkvT = bfp; bfp += (size_t)ML * MD * MD3;
    unsigned short* woutT = bfp; bfp += (size_t)ML * MD * MD;
    unsigned short* w1T   = bfp; bfp += (size_t)ML * MD * MF;
    unsigned short* w2T   = bfp; bfp += (size_t)ML * MF * MD;

    // one-time weight convert+transpose (weights reused 16x per layer; sit in L2)
    for (int l = 0; l < ML; ++l) {
        const float* Wqkv_l = ((const float*)d_in[4])  + (size_t)l * MD * MD3;
        const float* Wout_l = ((const float*)d_in[6])  + (size_t)l * MD * MD;
        const float* W1_l   = ((const float*)d_in[10]) + (size_t)l * MD * MF;
        const float* W2_l   = ((const float*)d_in[12]) + (size_t)l * MF * MD;
        transpose_bf16_kernel<<<(MD * MD3 + 255) / 256, 256, 0, stream>>>(
            Wqkv_l, wqkvT + (size_t)l * MD * MD3, MD, MD3);
        transpose_bf16_kernel<<<(MD * MD + 255) / 256, 256, 0, stream>>>(
            Wout_l, woutT + (size_t)l * MD * MD, MD, MD);
        transpose_bf16_kernel<<<(MD * MF + 255) / 256, 256, 0, stream>>>(
            W1_l, w1T + (size_t)l * MD * MF, MD, MF);
        transpose_bf16_kernel<<<(MF * MD + 255) / 256, 256, 0, stream>>>(
            W2_l, w2T + (size_t)l * MF * MD, MF, MD);
    }

    // embeddings -> d_out
    embed_kernel<<<MB * MT, 256, 0, stream>>>(input_ids, tok_emb, pos_emb, out);

    for (int l = 0; l < ML; ++l) {
        const float* bqkv_l  = ((const float*)d_in[5])  + (size_t)l * MD3;
        const float* bout_l  = ((const float*)d_in[7])  + (size_t)l * MD;
        const float* ln1s_l  = ((const float*)d_in[8])  + (size_t)l * MD;
        const float* ln1b_l  = ((const float*)d_in[9])  + (size_t)l * MD;
        const float* b1_l    = ((const float*)d_in[11]) + (size_t)l * MF;
        const float* b2_l    = ((const float*)d_in[13]) + (size_t)l * MD;
        const float* ln2s_l  = ((const float*)d_in[14]) + (size_t)l * MD;
        const float* ln2b_l  = ((const float*)d_in[15]) + (size_t)l * MD;
        const float* Wproj_l = ((const float*)d_in[16]) + (size_t)l * MD * MD;
        const float* Wattn_l = ((const float*)d_in[17]) + (size_t)l * MD * MD;
        const float* Wg1_l   = ((const float*)d_in[18]) + (size_t)l * MD3 * MU;
        const float* bg1_l   = ((const float*)d_in[19]) + (size_t)l * MU;
        const float* Wg2_l   = ((const float*)d_in[20]) + (size_t)l * MU * MD;
        const float* bg2_l   = ((const float*)d_in[21]) + (size_t)l * MD;
        const float* lngs_l  = ((const float*)d_in[22]) + (size_t)l * MD;
        const float* lngb_l  = ((const float*)d_in[23]) + (size_t)l * MD;

        const unsigned short* wqkvT_l = wqkvT + (size_t)l * MD * MD3;
        const unsigned short* woutT_l = woutT + (size_t)l * MD * MD;
        const unsigned short* w1T_l   = w1T   + (size_t)l * MD * MF;
        const unsigned short* w2T_l   = w2T   + (size_t)l * MF * MD;

        // scan init: h_spk = 0
        zero_kernel<<<(MS * MD + 255) / 256, 256, 0, stream>>>(h_spk, MS * MD);

        for (int b = 0; b < MB; ++b) {
            float* emb_b = out + (size_t)b * MT * MD;
            const int* ids_b = input_ids + (size_t)b * MT;

            // proj = h_spk[sid] @ Wproj ; x0 = emb_b + proj
            vecmat_kernel<0><<<(MD + 255) / 256, 256, 0, stream>>>(
                h_spk, Wproj_l, nullptr, proj, MD, MD, speaker_ids, b);
            add_rowvec_kernel<<<MT, 256, 0, stream>>>(emb_b, proj, x0);

            // qkv = x0 @ Wqkv + bqkv      (512x768 @ 768x2304)
            {
                dim3 g(MD3 / 64, MT / 128, 1);
                wmma_gemm_kernel<1, 2><<<g, 256, 0, stream>>>(
                    x0, MD, 0,  wqkvT_l, MD, 0,
                    qkv, MD3, 0, bqkv_l, nullptr, MD, 1.0f);
            }
            // scores[h] = (q_h @ k_h^T) / sqrt(HD)   (z over 12 heads)
            {
                dim3 g(MT / 64, MT / 128, MH);
                wmma_gemm_kernel<0, 1><<<g, 256, 0, stream>>>(
                    qkv, MD3, MHD,                      // A = q_h
                    qkv + MD, MD3, MHD,                 // B col n contiguous in k
                    scores, MT, (long)MT * MT,
                    nullptr, nullptr, MHD, 0.125f);
            }
            softmax_rows_kernel<<<MH * MT, 256, 0, stream>>>(scores, MT);
            // attnout[:, h*64:(h+1)*64] = scores_h @ v_h
            {
                dim3 g(MHD / 64, MT / 128, MH);
                wmma_gemm_kernel<0, 0><<<g, 256, 0, stream>>>(
                    scores, MT, (long)MT * MT,
                    qkv + 2 * MD, MD3, MHD,             // B = v_h (row-major k,n)
                    attnout, MD, MHD,
                    nullptr, nullptr, MT, 1.0f);
            }
            // tmp = x0 + attnout @ Wout + bout
            {
                dim3 g(MD / 64, MT / 128, 1);
                wmma_gemm_kernel<3, 2><<<g, 256, 0, stream>>>(
                    attnout, MD, 0,  woutT_l, MD, 0,
                    tmp, MD, 0,      bout_l, x0, MD, 1.0f);
            }
            layernorm_kernel<<<MT, 256, 0, stream>>>(tmp, ln1s_l, ln1b_l, x1);
            // hbuf = gelu(x1 @ W1 + b1)
            {
                dim3 g(MF / 64, MT / 128, 1);
                wmma_gemm_kernel<2, 2><<<g, 256, 0, stream>>>(
                    x1, MD, 0,  w1T_l, MD, 0,
                    hbuf, MF, 0, b1_l, nullptr, MD, 1.0f);
            }
            // tmp = x1 + hbuf @ W2 + b2
            {
                dim3 g(MD / 64, MT / 128, 1);
                wmma_gemm_kernel<3, 2><<<g, 256, 0, stream>>>(
                    hbuf, MF, 0,  w2T_l, MF, 0,
                    tmp, MD, 0,   b2_l, x1, MF, 1.0f);
            }
            // emb_b = LN2(tmp)   (this is the scan output x)
            layernorm_kernel<<<MT, 256, 0, stream>>>(tmp, ln2s_l, ln2b_l, emb_b);

            // ---- speaker-state update ----
            masked_mean_kernel<<<(MD + 255) / 256, 256, 0, stream>>>(emb_b, ids_b, u);
            vecmat_kernel<0><<<(MD + 255) / 256, 256, 0, stream>>>(
                u, Wattn_l, nullptr, uW, MD, MD, nullptr, 0);
            spk_attn_kernel<<<1, 256, 0, stream>>>(h_spk, uW, u, speaker_ids, b, mat);
            vecmat_kernel<1><<<(MU + 255) / 256, 256, 0, stream>>>(
                mat, Wg1_l, bg1_l, gbuf, MD3, MU, nullptr, 0);
            vecmat_kernel<0><<<(MD + 255) / 256, 256, 0, stream>>>(
                gbuf, Wg2_l, bg2_l, nh, MU, MD, nullptr, 0);
            ln_add_hspk_kernel<<<1, 256, 0, stream>>>(nh, lngs_l, lngb_l,
                                                      h_spk, speaker_ids, b);
        }
    }
}
